// SwinBlock_6322191859790
// MI455X (gfx1250) — matrix operand order repaired and checked by reading, not compiled, not run on hardware
//
#include <hip/hip_runtime.h>
#include <math.h>
#include <stdint.h>

// ---------------- types ----------------
typedef __attribute__((ext_vector_type(16))) __bf16 v16bf;
typedef __attribute__((ext_vector_type(8)))  float  v8f;
typedef __attribute__((ext_vector_type(4)))  unsigned int u32x4;
typedef __attribute__((ext_vector_type(8)))  unsigned int u32x8;

#define CDIM 384
#define FFDIM 1536
#define NHEAD 12
#define DH 32
#define LWIN 64
#define NWIN 2048     // 32 * (64/8) * (64/8)
#define NTOK 131072   // 32 * 64 * 64

__device__ __forceinline__ v8f wmma_bf16(v16bf a, v16bf b, v8f c) {
  return __builtin_amdgcn_wmma_f32_16x16x32_bf16(false, a, false, b, (short)0, c, false, false);
}
__device__ __forceinline__ float red4(float v) {
  v += __shfl_xor(v, 1, 4); v += __shfl_xor(v, 2, 4); return v;
}
__device__ __forceinline__ float red8(float v) {
  v += __shfl_xor(v, 1, 8); v += __shfl_xor(v, 2, 8); v += __shfl_xor(v, 4, 8); return v;
}
// WMMA A-fragment packing: for row m (within 16-row block) and ko = k % 32:
//   lane = m + 16*((ko>>4)&1), elem = ko & 15
__device__ __forceinline__ int frag_lane(int m16, int ko) { return m16 + ((ko & 16) ? 16 : 0); }
__device__ __forceinline__ int frag_elem(int ko)          { return ko & 15; }

// ---------------- kernel 0a: weights fp32 -> bf16 with LN-affine gamma folded ----------------
__global__ void k_fuse_w(const float* inproj, const float* outproj,
                         const float* w1, const float* w2,
                         const float* gq, const float* gk, const float* gv,
                         const float* g2, __bf16* dst) {
  const int N0 = 3 * CDIM * CDIM;
  const int N1 = CDIM * CDIM;
  const int N2 = FFDIM * CDIM;
  const int N3 = CDIM * FFDIM;
  int i = blockIdx.x * 256 + threadIdx.x;
  if (i >= N0 + N1 + N2 + N3) return;
  float v;
  if (i < N0) {
    int k = i % CDIM, p = i / (CDIM * CDIM);
    const float* g = (p == 0) ? gq : (p == 1) ? gk : gv;
    v = inproj[i] * g[k];
  } else if (i < N0 + N1) {
    v = outproj[i - N0];
  } else if (i < N0 + N1 + N2) {
    int j = i - N0 - N1;
    v = w1[j] * g2[j % CDIM];
  } else {
    v = w2[i - N0 - N1 - N2];
  }
  dst[i] = (__bf16)v;
}

// ---------------- kernel 0b: fused biases  bias' = bias + W @ beta ----------------
__global__ void k_fuse_b(const float* inproj_w, const float* inproj_b,
                         const float* bq, const float* bk, const float* bv,
                         const float* w1, const float* b1, const float* b2n,
                         float* fbias) {
  int i = blockIdx.x * 256 + threadIdx.x;
  if (i < 3 * CDIM) {
    int p = i / CDIM;
    const float* bp = (p == 0) ? bq : (p == 1) ? bk : bv;
    const float* wr = inproj_w + (size_t)i * CDIM;
    float s = inproj_b[i];
    for (int k = 0; k < CDIM; ++k) s += wr[k] * bp[k];
    fbias[i] = s;
  } else if (i < 3 * CDIM + FFDIM) {
    int n = i - 3 * CDIM;
    const float* wr = w1 + (size_t)n * CDIM;
    float s = b1[n];
    for (int k = 0; k < CDIM; ++k) s += wr[k] * b2n[k];
    fbias[i] = s;
  }
}

// ---------------- kernel 1: LN1 + window + QKV GEMMs ----------------
__global__ __launch_bounds__(256)
void k_qkv(const float* __restrict__ x,
           const float* g1, const float* b1,
           const __bf16* __restrict__ wbf, const float* __restrict__ fbias,
           float* __restrict__ win_ws,
           __bf16* __restrict__ qfrag, __bf16* __restrict__ k_ws, __bf16* __restrict__ vT) {
  __shared__ alignas(32) __bf16 afrag[4 * 12 * 32 * 16];   // 48 KB, fragment-packed
  const int win = blockIdx.x;
  const int bb_ = win >> 6, rem = win & 63, wh = rem >> 3, wx = rem & 7;
  const int tid = threadIdx.x, lane = tid & 31, wave = tid >> 5;

  // ---- LN phase: 4 threads per token row ----
  {
    const int tok = tid >> 2, part = tid & 3;
    const int r = tok >> 3, c = tok & 7;
    const float* xrow = x + (((size_t)(bb_ * 64 + wh * 8 + r) * 64) + (wx * 8 + c)) * CDIM;
    float* winrow = win_ws + ((size_t)win * LWIN + tok) * CDIM;
    float s = 0.f, s2 = 0.f;
    for (int j = part * 96; j < part * 96 + 96; ++j) { float t = xrow[j]; s += t; s2 += t * t; }
    s = red4(s); s2 = red4(s2);
    float m1 = s * (1.f / CDIM);
    float rs1 = rsqrtf(s2 * (1.f / CDIM) - m1 * m1 + 1e-5f);
    float u = 0.f, u2 = 0.f;
    for (int j = part * 96; j < part * 96 + 96; ++j) {
      float wv = (xrow[j] - m1) * rs1 * g1[j] + b1[j];
      winrow[j] = wv; u += wv; u2 += wv * wv;
    }
    u = red4(u); u2 = red4(u2);
    float m2 = u * (1.f / CDIM);
    float rs2 = rsqrtf(u2 * (1.f / CDIM) - m2 * m2 + 1e-5f);
    const int rbx = tok >> 4, m16 = tok & 15;
    for (int j = part * 96; j < part * 96 + 96; ++j) {
      int kt = j >> 5, ko = j & 31;
      afrag[((rbx * 12 + kt) * 32 + frag_lane(m16, ko)) * 16 + frag_elem(ko)] =
          (__bf16)((winrow[j] - m2) * rs2);
    }
  }
  __syncthreads();

  // ---- GEMM: 8 waves = 4 row-blocks x 2 col-halves (192 cols, 12 tiles) ----
  const int rb = wave & 3, ch = wave >> 2;
  const int khalf = lane >> 4;
  for (int p = 0; p < 3; ++p) {
    v8f acc[12];
    for (int t = 0; t < 12; ++t) acc[t] = (v8f){0.f,0.f,0.f,0.f,0.f,0.f,0.f,0.f};
    for (int kt = 0; kt < 12; ++kt) {
      const v16bf a = *(const v16bf*)&afrag[((rb * 12 + kt) * 32 + lane) * 16];
      #pragma unroll
      for (int nt = 0; nt < 12; ++nt) {
        int col = ch * 192 + nt * 16 + (lane & 15);
        const v16bf bm = *reinterpret_cast<const v16bf*>(
            wbf + ((size_t)(p * CDIM + col)) * CDIM + kt * 32 + khalf * 16);
        acc[nt] = wmma_bf16(a, bm, acc[nt]);
      }
    }
    #pragma unroll
    for (int nt = 0; nt < 12; ++nt) {
      int col = ch * 192 + nt * 16 + (lane & 15);
      float bias = fbias[p * CDIM + col];
      int head = col >> 5, d = col & 31;
      #pragma unroll
      for (int r8 = 0; r8 < 8; ++r8) {
        int row = rb * 16 + r8 + 8 * khalf;
        float v = acc[nt][r8] + bias;
        if (p == 0) {        // Q: fragment-packed [win][head][rb][lane][elem]
          qfrag[((((size_t)win * NHEAD + head) * 4 + rb) * 32 + frag_lane(row & 15, d)) * 16 +
                frag_elem(d)] = (__bf16)v;
        } else if (p == 1) { // K: row-major [win][head][tok][d] (B-operand of scores)
          k_ws[(((size_t)win * NHEAD + head) * LWIN + row) * DH + d] = (__bf16)v;
        } else {             // V: transposed [win][head][d][tok] (B-operand of P@V)
          vT[(((size_t)win * NHEAD + head) * DH + d) * LWIN + row] = (__bf16)v;
        }
      }
    }
  }
}

// ---------------- kernel 2: windowed attention (per window, per head) ----------------
__global__ __launch_bounds__(128)
void k_attn(const __bf16* __restrict__ qfrag, const __bf16* __restrict__ k_ws,
            const __bf16* __restrict__ vT, __bf16* __restrict__ ofrag) {
  __shared__ float  s_lds[LWIN * 68];
  __shared__ alignas(32) __bf16 pfrag[4 * 2 * 32 * 16];   // P fragment-packed (8 KB)
  const int win = blockIdx.x, head = blockIdx.y;
  const int tid = threadIdx.x, lane = tid & 31, wave = tid >> 5;
  const int khalf = lane >> 4;
  const __bf16* kb_ = k_ws + ((size_t)win * NHEAD + head) * LWIN * DH;
  const __bf16* vb_ = vT + ((size_t)win * NHEAD + head) * DH * LWIN;

  // scores: row-block per wave, K=32 single WMMA per tile
  {
    const v16bf aq = *(const v16bf*)(qfrag +
        ((((size_t)win * NHEAD + head) * 4 + wave) * 32 + lane) * 16);
    #pragma unroll
    for (int nt = 0; nt < 4; ++nt) {
      int n = nt * 16 + (lane & 15);
      const v16bf bm = *reinterpret_cast<const v16bf*>(kb_ + n * DH + khalf * 16);
      v8f z = (v8f){0.f,0.f,0.f,0.f,0.f,0.f,0.f,0.f};
      v8f sc = wmma_bf16(aq, bm, z);
      #pragma unroll
      for (int r8 = 0; r8 < 8; ++r8)
        s_lds[(wave * 16 + r8 + 8 * khalf) * 68 + nt * 16 + (lane & 15)] = sc[r8];
    }
  }
  __syncthreads();

  // softmax: 2 threads per row; write P in fragment-packed layout
  {
    const int row = tid >> 1, halfr = tid & 1;
    const float scale = 0.17677669529663687f; // 1/sqrt(32)
    float vals[32], mx = -1e30f;
    #pragma unroll
    for (int j = 0; j < 32; ++j) {
      float t = s_lds[row * 68 + halfr * 32 + j] * scale;
      vals[j] = t; mx = fmaxf(mx, t);
    }
    mx = fmaxf(mx, __shfl_xor(mx, 1, 2));
    float sum = 0.f;
    #pragma unroll
    for (int j = 0; j < 32; ++j) { vals[j] = __expf(vals[j] - mx); sum += vals[j]; }
    sum += __shfl_xor(sum, 1, 2);
    float inv = 1.f / sum;
    const int rbp = row >> 4, m16 = row & 15;
    #pragma unroll
    for (int j = 0; j < 32; ++j)
      pfrag[((rbp * 2 + halfr) * 32 + frag_lane(m16, j)) * 16 + frag_elem(j)] =
          (__bf16)(vals[j] * inv);
  }
  __syncthreads();

  // O = P @ V : 16x32 per wave, K=64
  {
    v8f oacc[2];
    oacc[0] = (v8f){0.f,0.f,0.f,0.f,0.f,0.f,0.f,0.f};
    oacc[1] = oacc[0];
    for (int kt = 0; kt < 2; ++kt) {
      const v16bf ap = *(const v16bf*)&pfrag[((wave * 2 + kt) * 32 + lane) * 16];
      #pragma unroll
      for (int nt = 0; nt < 2; ++nt) {
        int n = nt * 16 + (lane & 15);          // d index
        const v16bf bm = *reinterpret_cast<const v16bf*>(
            vb_ + (size_t)n * LWIN + kt * 32 + khalf * 16);
        oacc[nt] = wmma_bf16(ap, bm, oacc[nt]);
      }
    }
    // O fragment-packed: [win][rb=wave][kt=head][lane][elem]
    #pragma unroll
    for (int nt = 0; nt < 2; ++nt) {
      int d = nt * 16 + (lane & 15);
      #pragma unroll
      for (int r8 = 0; r8 < 8; ++r8) {
        int m16 = r8 + 8 * khalf;
        ofrag[((((size_t)win * 4 + wave) * NHEAD + head) * 32 + frag_lane(m16, d)) * 16 +
              frag_elem(d)] = (__bf16)oacc[nt][r8];
      }
    }
  }
}

// ---------------- kernel 3: out-proj + residual + nout LN + unwindow ----------------
#define LDA 392
__global__ __launch_bounds__(256)
void k_outproj(const __bf16* __restrict__ ofrag, const __bf16* __restrict__ woutbf,
               const float* __restrict__ outb, const float* __restrict__ win_ws,
               const float* __restrict__ x, const float* gout, const float* bout,
               float* __restrict__ y_ws) {
  __shared__ float t_lds[32 * LDA];
  const int blk = blockIdx.x, win = blk >> 1, halfw = blk & 1;
  const int tid = threadIdx.x, lane = tid & 31, wave = tid >> 5;
  const int rb = wave & 1, ch = wave >> 1;   // 2 row-blocks x 4 col-groups (96 cols)
  const int khalf = lane >> 4;
  const int rbo = halfw * 2 + rb;
  v8f acc[6];
  for (int t = 0; t < 6; ++t) acc[t] = (v8f){0.f,0.f,0.f,0.f,0.f,0.f,0.f,0.f};
  for (int kt = 0; kt < 12; ++kt) {
    const v16bf a = *(const v16bf*)(ofrag +
        ((((size_t)win * 4 + rbo) * 12 + kt) * 32 + lane) * 16);
    #pragma unroll
    for (int nt = 0; nt < 6; ++nt) {
      int col = ch * 96 + nt * 16 + (lane & 15);
      const v16bf bm = *reinterpret_cast<const v16bf*>(
          woutbf + (size_t)col * CDIM + kt * 32 + khalf * 16);
      acc[nt] = wmma_bf16(a, bm, acc[nt]);
    }
  }
  #pragma unroll
  for (int nt = 0; nt < 6; ++nt) {
    int col = ch * 96 + nt * 16 + (lane & 15);
    float bias = outb[col];
    #pragma unroll
    for (int r8 = 0; r8 < 8; ++r8) {
      int lrow = rb * 16 + r8 + 8 * khalf;
      int grow = halfw * 32 + lrow;
      t_lds[lrow * LDA + col] =
          acc[nt][r8] + bias + win_ws[((size_t)win * LWIN + grow) * CDIM + col];
    }
  }
  __syncthreads();
  // LN per row (8 threads/row) + unwindow residual into y
  {
    const int lrow = tid >> 3, part = tid & 7;
    float s = 0.f, s2 = 0.f;
    for (int j = part * 48; j < part * 48 + 48; ++j) {
      float t = t_lds[lrow * LDA + j]; s += t; s2 += t * t;
    }
    s = red8(s); s2 = red8(s2);
    float m = s * (1.f / CDIM);
    float rs = rsqrtf(s2 * (1.f / CDIM) - m * m + 1e-5f);
    const int l = halfw * 32 + lrow;
    const int bb_ = win >> 6, rem = win & 63, wh = rem >> 3, wx = rem & 7;
    const size_t gidx =
        (((size_t)(bb_ * 64 + wh * 8 + (l >> 3))) * 64 + wx * 8 + (l & 7)) * CDIM;
    for (int j = part * 48; j < part * 48 + 48; ++j) {
      float t = t_lds[lrow * LDA + j];
      y_ws[gidx + j] = x[gidx + j] + (t - m) * rs * gout[j] + bout[j];
    }
  }
}

// ---------------- kernel 4: LN2 + FFN up + GELU (h1 written fragment-packed) ----------------
__global__ __launch_bounds__(256)
void k_ffn1(const float* __restrict__ y_ws,
            const __bf16* __restrict__ w1bf, const float* __restrict__ fbias,
            __bf16* __restrict__ h1f) {
  __shared__ alignas(32) __bf16 afrag[4 * 12 * 32 * 16];
  const int blk = blockIdx.x;
  const int rbase = blk * 64;
  const int tid = threadIdx.x, lane = tid & 31, wave = tid >> 5;
  {
    const int tok = tid >> 2, part = tid & 3;
    const float* yrow = y_ws + (size_t)(rbase + tok) * CDIM;
    float s = 0.f, s2 = 0.f;
    for (int j = part * 96; j < part * 96 + 96; ++j) { float t = yrow[j]; s += t; s2 += t * t; }
    s = red4(s); s2 = red4(s2);
    float m = s * (1.f / CDIM);
    float rs = rsqrtf(s2 * (1.f / CDIM) - m * m + 1e-5f);
    const int rbx = tok >> 4, m16 = tok & 15;
    for (int j = part * 96; j < part * 96 + 96; ++j) {   // gamma/beta folded into w1/fbias
      int kt = j >> 5, ko = j & 31;
      afrag[((rbx * 12 + kt) * 32 + frag_lane(m16, ko)) * 16 + frag_elem(ko)] =
          (__bf16)((yrow[j] - m) * rs);
    }
  }
  __syncthreads();
  const int rb = wave & 3, ch = wave >> 2;
  const int khalf = lane >> 4;
  for (int oc = 0; oc < 4; ++oc) {
    const int cbase = (oc * 2 + ch) * 192;
    v8f acc[12];
    for (int t = 0; t < 12; ++t) acc[t] = (v8f){0.f,0.f,0.f,0.f,0.f,0.f,0.f,0.f};
    for (int kt = 0; kt < 12; ++kt) {
      const v16bf a = *(const v16bf*)&afrag[((rb * 12 + kt) * 32 + lane) * 16];
      #pragma unroll
      for (int nt = 0; nt < 12; ++nt) {
        int col = cbase + nt * 16 + (lane & 15);
        const v16bf bm = *reinterpret_cast<const v16bf*>(
            w1bf + (size_t)col * CDIM + kt * 32 + khalf * 16);
        acc[nt] = wmma_bf16(a, bm, acc[nt]);
      }
    }
    #pragma unroll
    for (int nt = 0; nt < 12; ++nt) {
      int col = cbase + nt * 16 + (lane & 15);
      float bias = fbias[3 * CDIM + col];
      int kt2 = col >> 5, ko = col & 31;
      #pragma unroll
      for (int r8 = 0; r8 < 8; ++r8) {
        int m16 = r8 + 8 * khalf;
        float t = acc[nt][r8] + bias;
        float ge = 0.5f * t * (1.f + erff(t * 0.70710678118f)); // exact GELU
        // fragment-packed: [blk][kt(48)][rb(4)][lane][elem] -> contiguous 4 KB per (blk,kt)
        h1f[((((size_t)blk * 48 + kt2) * 4 + rb) * 32 + frag_lane(m16, ko)) * 16 +
            frag_elem(ko)] = (__bf16)ge;
      }
    }
  }
}

// ---------------- kernel 5: FFN down + residual; A staged via TDM double-buffer ----------------
__global__ __launch_bounds__(256)
void k_ffn2(const __bf16* __restrict__ h1f, const __bf16* __restrict__ w2bf,
            const float* __restrict__ b2f, const float* __restrict__ y_ws,
            float* __restrict__ out) {
  __shared__ alignas(128) __bf16 stage[2][2048];   // 2 x 4 KB fragment chunks
  const int blk = blockIdx.x;
  const int rbase = blk * 64;
  const int tid = threadIdx.x, lane = tid & 31, wave = tid >> 5;
  const int rb = wave & 3, ch = wave >> 2;
  const int khalf = lane >> 4;
  const uint32_t lds0 = (uint32_t)(uintptr_t)(&stage[0][0]);
  const uint64_t gbase = (uint64_t)(uintptr_t)h1f + (uint64_t)blk * 48 * 4096;

  auto tdm_issue = [&](int kt, int buf) {
    uint64_t ga = gbase + (uint64_t)kt * 4096;
    u32x4 g0;
    g0[0] = 1u;                                                   // count=1 valid descriptor
    g0[1] = lds0 + (uint32_t)buf * 4096u;                         // lds_addr
    g0[2] = (uint32_t)ga;                                         // global_addr[31:0]
    g0[3] = ((uint32_t)(ga >> 32) & 0x01FFFFFFu) | 0x80000000u;   // addr[56:32] | type=2
    u32x8 g1;
    g1[0] = 0x00030000u;            // workgroup_mask=0, data_size=3 (8B units)
    g1[1] = 0u;                     // tensor_dim0[15:0]=0 (dim0 = 1<<20)
    g1[2] = 16u | (1u << 16);       // tensor_dim0[31:16]=16, tensor_dim1[15:0]=1
    g1[3] = (512u << 16);           // tensor_dim1 hi=0, tile_dim0=512 (4 KB line)
    g1[4] = 1u;                     // tile_dim1=1, tile_dim2=0
    g1[5] = 512u;                   // tensor_dim0_stride
    g1[6] = 0u;
    g1[7] = 0u;
    asm volatile("tensor_load_to_lds %0, %1" :: "s"(g0), "s"(g1) : "memory");
  };

  v8f acc[12];
  for (int t = 0; t < 12; ++t) acc[t] = (v8f){0.f,0.f,0.f,0.f,0.f,0.f,0.f,0.f};

  if (wave == 0) {
    tdm_issue(0, 0);
    __builtin_amdgcn_s_wait_tensorcnt(0);
  }
  __syncthreads();

  for (int kt = 0; kt < 48; ++kt) {
    if (wave == 0 && kt + 1 < 48) tdm_issue(kt + 1, (kt + 1) & 1);
    const v16bf a = *(const v16bf*)&stage[kt & 1][(rb * 32 + lane) * 16];
    #pragma unroll
    for (int nt = 0; nt < 12; ++nt) {
      int col = ch * 192 + nt * 16 + (lane & 15);
      const v16bf bm = *reinterpret_cast<const v16bf*>(
          w2bf + (size_t)col * FFDIM + kt * 32 + khalf * 16);
      acc[nt] = wmma_bf16(a, bm, acc[nt]);
    }
    if (wave == 0) {
      if (kt + 1 < 48) __builtin_amdgcn_s_wait_tensorcnt(1);
      else             __builtin_amdgcn_s_wait_tensorcnt(0);
    }
    __syncthreads();
  }

  #pragma unroll
  for (int nt = 0; nt < 12; ++nt) {
    int col = ch * 192 + nt * 16 + (lane & 15);
    float bias = b2f[col];
    #pragma unroll
    for (int r8 = 0; r8 < 8; ++r8) {
      size_t row = rbase + rb * 16 + r8 + 8 * khalf;
      out[row * CDIM + col] = y_ws[row * CDIM + col] + acc[nt][r8] + bias;
    }
  }
}

// ---------------- launch ----------------
extern "C" void kernel_launch(void* const* d_in, const int* in_sizes, int n_in,
                              void* d_out, int out_size, void* d_ws, size_t ws_size,
                              hipStream_t stream) {
  const float* x          = (const float*)d_in[0];
  const float* norm1_g    = (const float*)d_in[1];
  const float* norm1_b    = (const float*)d_in[2];
  const float* nq_g       = (const float*)d_in[3];
  const float* nq_b       = (const float*)d_in[4];
  const float* nk_g       = (const float*)d_in[5];
  const float* nk_b       = (const float*)d_in[6];
  const float* nv_g       = (const float*)d_in[7];
  const float* nv_b       = (const float*)d_in[8];
  const float* nout_g     = (const float*)d_in[9];
  const float* nout_b     = (const float*)d_in[10];
  const float* in_proj_w  = (const float*)d_in[11];
  const float* in_proj_b  = (const float*)d_in[12];
  const float* out_proj_w = (const float*)d_in[13];
  const float* out_proj_b = (const float*)d_in[14];
  const float* norm2_g    = (const float*)d_in[15];
  const float* norm2_b    = (const float*)d_in[16];
  const float* w1         = (const float*)d_in[17];
  const float* b1         = (const float*)d_in[18];
  const float* w2         = (const float*)d_in[19];
  const float* b2         = (const float*)d_in[20];
  float* out = (float*)d_out;

  // workspace layout (bytes)
  char* ws = (char*)d_ws;
  const size_t WB_INPROJ  = 0;                          // 884736 B (gamma-fused, bf16)
  const size_t WB_OUTPROJ = 884736;                     // 294912 B
  const size_t WB_W1      = 1179648;                    // 1179648 B (gamma-fused)
  const size_t WB_W2      = 2359296;                    // 1179648 B
  const size_t FB_OFF     = 3538944;                    // 2688 floats fused biases
  const size_t WIN_OFF    = 4194304;                    // 201326592 B fp32 windowed LN1(x)
  const size_t Q_OFF      = WIN_OFF + 201326592;        // 100663296 B bf16 (frag-packed)
  const size_t K_OFF      = Q_OFF + 100663296;          // row-major
  const size_t V_OFF      = K_OFF + 100663296;          // transposed
  const size_t O_OFF      = V_OFF + 100663296;          // frag-packed
  const size_t H1_OFF     = Q_OFF;                      // aliases q/k/v/o (402653184 B)
  const size_t Y_OFF      = O_OFF + 100663296;          // 201326592 B fp32

  __bf16* wbf    = (__bf16*)(ws + WB_INPROJ);
  __bf16* woutbf = (__bf16*)(ws + WB_OUTPROJ);
  __bf16* w1bf   = (__bf16*)(ws + WB_W1);
  __bf16* w2bf   = (__bf16*)(ws + WB_W2);
  float*  fbias  = (float*)(ws + FB_OFF);
  float*  win_ws = (float*)(ws + WIN_OFF);
  __bf16* qfrag  = (__bf16*)(ws + Q_OFF);
  __bf16* k_ws   = (__bf16*)(ws + K_OFF);
  __bf16* vT     = (__bf16*)(ws + V_OFF);
  __bf16* ofrag  = (__bf16*)(ws + O_OFF);
  __bf16* h1f    = (__bf16*)(ws + H1_OFF);
  float*  y_ws   = (float*)(ws + Y_OFF);

  const int wtotal = 3*CDIM*CDIM + CDIM*CDIM + FFDIM*CDIM + CDIM*FFDIM;
  k_fuse_w<<<(wtotal + 255) / 256, 256, 0, stream>>>(in_proj_w, out_proj_w, w1, w2,
                                                     nq_g, nk_g, nv_g, norm2_g, wbf);
  k_fuse_b<<<(3*CDIM + FFDIM + 255) / 256, 256, 0, stream>>>(
      in_proj_w, in_proj_b, nq_b, nk_b, nv_b, w1, b1, norm2_b, fbias);

  k_qkv<<<NWIN, 256, 0, stream>>>(x, norm1_g, norm1_b, wbf, fbias,
                                  win_ws, qfrag, k_ws, vT);

  k_attn<<<dim3(NWIN, NHEAD), 128, 0, stream>>>(qfrag, k_ws, vT, ofrag);

  k_outproj<<<NWIN * 2, 256, 0, stream>>>(ofrag, woutbf, out_proj_b, win_ws, x,
                                          nout_g, nout_b, y_ws);

  k_ffn1<<<NTOK / 64, 256, 0, stream>>>(y_ws, w1bf, fbias, h1f);

  k_ffn2<<<NTOK / 64, 256, 0, stream>>>(h1f, w2bf, b2, y_ws, out);
}
